// GCN_90237262889600
// MI455X (gfx1250) — matrix operand order repaired and checked by reading, not compiled, not run on hardware
//
#include <hip/hip_runtime.h>
#include <hip/hip_bf16.h>

typedef __attribute__((ext_vector_type(16))) __bf16 v16bf;
typedef __attribute__((ext_vector_type(8)))  float  v8f;

#define GD 384           // feature dim
#define GOUT 2

// ---------------- utility: zero fill ----------------
__global__ void gcn_fill0(float* __restrict__ p, size_t n) {
    size_t i = (size_t)blockIdx.x * blockDim.x + threadIdx.x;
    size_t stride = (size_t)gridDim.x * blockDim.x;
    for (; i < n; i += stride) p[i] = 0.0f;
}

// ---------------- in-degree count (float for later divide) ----------------
__global__ void gcn_count(const int* __restrict__ dst, float* __restrict__ cnt, int E) {
    int e = blockIdx.x * blockDim.x + threadIdx.x;
    if (e < E) unsafeAtomicAdd(&cnt[dst[e]], 1.0f);
}

// ---------------- edge scatter: agg[dst] += w * x[src] ----------------
__global__ void gcn_scatter(const float* __restrict__ xin,
                            const int* __restrict__ src, const int* __restrict__ dst,
                            const float* __restrict__ w, float* __restrict__ agg, int E) {
    int e = blockIdx.x;
    if (e >= E) return;
    int s = src[e], d = dst[e];
    float wt = w[e];
    const float* xr = xin + (size_t)s * GD;
    float*       ar = agg + (size_t)d * GD;
    int k = threadIdx.x;                   // blockDim.x == 384, one pass
    unsafeAtomicAdd(&ar[k], wt * xr[k]);
}

// ---------------- fused dual-GEMM + bias + ReLU via bf16 WMMA ----------------
// out[m,n] = relu( (Aagg[m,:]/max(cnt[m],1)) @ Wrel[:,n] + brel[n] + Aroot[m,:] @ Wroot[:,n] )
// block: 256 threads = 8 waves; tile = 128 rows x 16 cols; K loop = 384/32 = 12 steps
// Two independent accumulator chains (rel / root) so back-to-back WMMAs have no
// D->C RAW hazard; chains are summed once in the epilogue.
__global__ __launch_bounds__(256)
void gcn_gemm_wmma(const float* __restrict__ Aagg, const float* __restrict__ cnt,
                   const float* __restrict__ Aroot,
                   const float* __restrict__ Wrel, const float* __restrict__ brel,
                   const float* __restrict__ Wroot,
                   float* __restrict__ out, int M) {
    // LDS: weight columns n0..n0+15 in N-major bf16 layout -> B fragment = contiguous 32B
    __shared__ __align__(32) __bf16 sWrel[16 * GD];
    __shared__ __align__(32) __bf16 sWroot[16 * GD];

    const int n0  = blockIdx.y * 16;
    const int m0  = blockIdx.x * 128;
    const int tid = threadIdx.x;

    for (int i = tid; i < 16 * GD; i += 256) {
        int n = i & 15;          // consecutive tids read consecutive global floats
        int k = i >> 4;
        sWrel[n * GD + k]  = (__bf16)Wrel[(size_t)k * GD + n0 + n];
        sWroot[n * GD + k] = (__bf16)Wroot[(size_t)k * GD + n0 + n];
    }
    __syncthreads();

    const int wave = tid >> 5;       // 0..7
    const int lane = tid & 31;
    const int hsel = lane >> 4;      // 0/1 : which K-half this lane carries
    const int lr   = lane & 15;
    const int row  = m0 + wave * 16 + lr;    // A row carried by this lane
    const int arow = (row < M) ? row : 0;    // clamp: keep EXEC all-ones for WMMA

    const float c     = cnt[arow];
    const float scale = 1.0f / fmaxf(c, 1.0f);

    const float* aggRow  = Aagg  + (size_t)arow * GD;
    const float* rootRow = Aroot + (size_t)arow * GD;

    v8f accRel, accRoot;
    const float bv = brel[n0 + lr];          // bias depends only on column
    #pragma unroll
    for (int j = 0; j < 8; ++j) { accRel[j] = bv; accRoot[j] = 0.0f; }

    #pragma unroll 2
    for (int kk = 0; kk < GD; kk += 32) {
        // ---- A fragments (contiguous K order, matching B):
        // lane element j(0..7)  -> K = kk + hsel*16 + j
        // lane element j(8..15) -> K = kk + hsel*16 + j
        const float4* pa0 = (const float4*)(aggRow  + kk + hsel * 16);
        const float4* pa1 = (const float4*)(aggRow  + kk + hsel * 16 + 8);
        const float4* pr0 = (const float4*)(rootRow + kk + hsel * 16);
        const float4* pr1 = (const float4*)(rootRow + kk + hsel * 16 + 8);
        float4 a0 = pa0[0], a1q = pa0[1], a2 = pa1[0], a3 = pa1[1];
        float4 r0 = pr0[0], r1q = pr0[1], r2 = pr1[0], r3 = pr1[1];
        float af[16] = { a0.x,a0.y,a0.z,a0.w, a1q.x,a1q.y,a1q.z,a1q.w,
                         a2.x,a2.y,a2.z,a2.w, a3.x,a3.y,a3.z,a3.w };
        float rf[16] = { r0.x,r0.y,r0.z,r0.w, r1q.x,r1q.y,r1q.z,r1q.w,
                         r2.x,r2.y,r2.z,r2.w, r3.x,r3.y,r3.z,r3.w };
        v16bf aAgg, aRoot;
        #pragma unroll
        for (int j = 0; j < 16; ++j) {
            aAgg[j]  = (__bf16)(af[j] * scale);   // fold mean-divide into A
            aRoot[j] = (__bf16)rf[j];
        }

        // ---- B fragments: lane col = lr; elements 2v,2v+1 = K = kk + hsel*16 + 2v(+1)
        // -> 16 contiguous bf16 in LDS (32B, aligned)
        v16bf bRel  = *(const v16bf*)&sWrel[lr * GD + kk + hsel * 16];
        v16bf bRoot = *(const v16bf*)&sWroot[lr * GD + kk + hsel * 16];

        accRel  = __builtin_amdgcn_wmma_f32_16x16x32_bf16(false, aAgg,  false, bRel,
                                                          (short)0, accRel,  false, false);
        accRoot = __builtin_amdgcn_wmma_f32_16x16x32_bf16(false, aRoot, false, bRoot,
                                                          (short)0, accRoot, false, false);
    }

    // C/D layout: lanes 0-15: VGPR j -> M=j ; lanes 16-31: VGPR j -> M=j+8 ; N = lr
    const int rbase = m0 + wave * 16 + hsel * 8;     // first row this lane stores
    if (rbase + 7 < M) {
        // fast path: whole 8-row group valid (always true when M % 16 == 0)
        float* orow = out + (size_t)rbase * GD + n0 + lr;
        #pragma unroll
        for (int j = 0; j < 8; ++j) {
            float v = accRel[j] + accRoot[j];
            orow[(size_t)j * GD] = v > 0.0f ? v : 0.0f;
        }
    } else {
        #pragma unroll
        for (int j = 0; j < 8; ++j) {
            int r = rbase + j;
            if (r < M) {
                float v = accRel[j] + accRoot[j];
                out[(size_t)r * GD + n0 + lr] = v > 0.0f ? v : 0.0f;
            }
        }
    }
}

// ---------------- sigmoid head: wave per node ----------------
__global__ __launch_bounds__(256)
void gcn_head(const float* __restrict__ y, const float* __restrict__ Wl,
              const float* __restrict__ bl, float* __restrict__ out, int M) {
    int node = blockIdx.x * (blockDim.x >> 5) + (threadIdx.x >> 5);
    int lane = threadIdx.x & 31;
    if (node >= M) return;
    const float* yr = y + (size_t)node * GD;
    float s0 = 0.0f, s1 = 0.0f;
    for (int k = lane; k < GD; k += 32) {
        float v = yr[k];
        s0 += v * Wl[k * GOUT + 0];
        s1 += v * Wl[k * GOUT + 1];
    }
    #pragma unroll
    for (int off = 16; off > 0; off >>= 1) {
        s0 += __shfl_down(s0, off, 32);
        s1 += __shfl_down(s1, off, 32);
    }
    if (lane == 0) {
        out[(size_t)node * GOUT + 0] = 1.0f / (1.0f + __expf(-(s0 + bl[0])));
        out[(size_t)node * GOUT + 1] = 1.0f / (1.0f + __expf(-(s1 + bl[1])));
    }
}

extern "C" void kernel_launch(void* const* d_in, const int* in_sizes, int n_in,
                              void* d_out, int out_size, void* d_ws, size_t ws_size,
                              hipStream_t stream) {
    (void)n_in; (void)out_size; (void)ws_size;
    const float* x   = (const float*)d_in[0];
    const int*   ei  = (const int*)  d_in[1];
    const float* ew  = (const float*)d_in[2];
    const float* W1r = (const float*)d_in[3];
    const float* b1r = (const float*)d_in[4];
    const float* W1o = (const float*)d_in[5];
    const float* W2r = (const float*)d_in[6];
    const float* b2r = (const float*)d_in[7];
    const float* W2o = (const float*)d_in[8];
    const float* Wl  = (const float*)d_in[9];
    const float* bl  = (const float*)d_in[10];

    const int N = in_sizes[0] / GD;       // 50000
    const int E = in_sizes[1] / 2;        // 200000
    const int* src = ei;
    const int* dst = ei + E;

    float* agg  = (float*)d_ws;                       // [N, 384]
    float* h    = agg + (size_t)N * GD;               // [N, 384]
    float* cnt  = h   + (size_t)N * GD;               // [N]
    float* outp = (float*)d_out;                      // [N, 2]
    float* y    = outp + (size_t)N * GOUT;            // [N, 384] (2nd tuple output)

    dim3 gemmGrid((N + 127) / 128, GD / 16);          // 391 x 24

    // ---- layer 1 ----
    gcn_fill0<<<2048, 256, 0, stream>>>(agg, (size_t)N * GD);
    gcn_fill0<<<64,   256, 0, stream>>>(cnt, (size_t)N);
    gcn_count<<<(E + 255) / 256, 256, 0, stream>>>(dst, cnt, E);
    gcn_scatter<<<E, GD, 0, stream>>>(x, src, dst, ew, agg, E);
    gcn_gemm_wmma<<<gemmGrid, 256, 0, stream>>>(agg, cnt, x, W1r, b1r, W1o, h, N);

    // ---- layer 2 (reuse agg; cnt identical) ----
    gcn_fill0<<<2048, 256, 0, stream>>>(agg, (size_t)N * GD);
    gcn_scatter<<<E, GD, 0, stream>>>(h, src, dst, ew, agg, E);
    gcn_gemm_wmma<<<gemmGrid, 256, 0, stream>>>(agg, cnt, h, W2r, b2r, W2o, y, N);

    // ---- head ----
    gcn_head<<<(N + 7) / 8, 256, 0, stream>>>(y, Wl, bl, outp, N);
}
